// MultiHeadAttention_55482387529983
// MI455X (gfx1250) — matrix operand order repaired and checked by reading, not compiled, not run on hardware
//
#include <hip/hip_runtime.h>

typedef __bf16 v16bf __attribute__((ext_vector_type(16)));
typedef __bf16 bf8v  __attribute__((ext_vector_type(8)));
typedef float  v8f   __attribute__((ext_vector_type(8)));

constexpr int kD  = 1024;    // d_model
constexpr int kH  = 16;      // heads
constexpr int kDk = 64;      // head dim
constexpr int kB  = 2;       // batch
constexpr int kS  = 2048;    // sequence
constexpr int kM  = kB * kS; // 4096 tokens

static __device__ inline v8f v8zero() {
  v8f z = {0.f, 0.f, 0.f, 0.f, 0.f, 0.f, 0.f, 0.f};
  return z;
}

static __device__ inline v8f wmma_bf16(v16bf a, v16bf b, v8f c) {
  return __builtin_amdgcn_wmma_f32_16x16x32_bf16(
      false, a, false, b, (short)0, c, false, false);
}

// Raw v_max_num_f32: same op as fmaxf but without the compiler's canonicalize
// (max(x,x)) preamble on intrinsic/asm-sourced operands.
static __device__ inline float fmax_raw(float a, float b) {
  float r;
  asm("v_max_num_f32 %0, %1, %2" : "=v"(r) : "v"(a), "v"(b));
  return r;
}

// ---------------------------------------------------------------------------
// Fragment loaders (all per-lane contiguous -> b128 loads)
// A 16x32 frag = two 8-elem runs per lane: K in [8h, 8h+8) and [16+8h, 16+8h+8)
// ---------------------------------------------------------------------------
static __device__ inline v16bf load_a_frag(const __bf16* __restrict__ src, int ld, int lane) {
  const int half = lane >> 4;
  const int row  = lane & 15;
  const bf8v lo = *(const bf8v*)(src + row * ld + 8 * half);
  const bf8v hi = *(const bf8v*)(src + row * ld + 16 + 8 * half);
  return __builtin_shufflevector(lo, hi, 0, 1, 2, 3, 4, 5, 6, 7,
                                 8, 9, 10, 11, 12, 13, 14, 15);
}

// B 32x16 frag, element (kk,n) at src[n*nstride + kk] (K contiguous per lane):
// lane n reads two 8-elem runs K in [8h, 8h+8) and [16+8h, 16+8h+8).
static __device__ inline v16bf load_b_frag_kcontig(const __bf16* __restrict__ src,
                                                   int nstride, int lane) {
  const int half = lane >> 4;
  const int n    = lane & 15;
  const __bf16* p = src + (size_t)n * nstride + 8 * half;
  const bf8v lo = *(const bf8v*)p;
  const bf8v hi = *(const bf8v*)(p + 16);
  return __builtin_shufflevector(lo, hi, 0, 1, 2, 3, 4, 5, 6, 7,
                                 8, 9, 10, 11, 12, 13, 14, 15);
}

// ---------------------------------------------------------------------------
// 16-lane row reductions: single-instruction DPP folds (dst = op(dpp(s0), s1)).
// Masks {1,3,7,15} span GF(2)^4; all patterns stay inside a 16-lane DPP row,
// so the two wave halves (holding different matrix rows) reduce independently.
// ---------------------------------------------------------------------------
static __device__ inline float red_max16(float v) {
  asm("v_max_num_f32_dpp %0, %0, %0 quad_perm:[1,0,3,2] row_mask:0xf bank_mask:0xf bound_ctrl:1"
      : "+v"(v));
  asm("v_max_num_f32_dpp %0, %0, %0 quad_perm:[3,2,1,0] row_mask:0xf bank_mask:0xf bound_ctrl:1"
      : "+v"(v));
  asm("v_max_num_f32_dpp %0, %0, %0 row_half_mirror row_mask:0xf bank_mask:0xf bound_ctrl:1"
      : "+v"(v));
  asm("v_max_num_f32_dpp %0, %0, %0 row_mirror row_mask:0xf bank_mask:0xf bound_ctrl:1"
      : "+v"(v));
  return v;
}
static __device__ inline float red_sum16(float v) {
  asm("v_add_f32_dpp %0, %0, %0 quad_perm:[1,0,3,2] row_mask:0xf bank_mask:0xf bound_ctrl:1"
      : "+v"(v));
  asm("v_add_f32_dpp %0, %0, %0 quad_perm:[3,2,1,0] row_mask:0xf bank_mask:0xf bound_ctrl:1"
      : "+v"(v));
  asm("v_add_f32_dpp %0, %0, %0 row_half_mirror row_mask:0xf bank_mask:0xf bound_ctrl:1"
      : "+v"(v));
  asm("v_add_f32_dpp %0, %0, %0 row_mirror row_mask:0xf bank_mask:0xf bound_ctrl:1"
      : "+v"(v));
  return v;
}

// ---------------------------------------------------------------------------
// Elementwise fp32 -> bf16 (one-time conversion of activations and weights)
// ---------------------------------------------------------------------------
__global__ void __launch_bounds__(256)
cvt_bf16_kernel(const float* __restrict__ in, __bf16* __restrict__ out, int n) {
  const int i = (blockIdx.x * 256 + threadIdx.x) * 8;
  if (i >= n) return;
  const float4 a = *(const float4*)(in + i);
  const float4 b = *(const float4*)(in + i + 4);
  bf8v r;
  r[0] = (__bf16)a.x; r[1] = (__bf16)a.y; r[2] = (__bf16)a.z; r[3] = (__bf16)a.w;
  r[4] = (__bf16)b.x; r[5] = (__bf16)b.y; r[6] = (__bf16)b.z; r[7] = (__bf16)b.w;
  *(bf8v*)(out + i) = r;
}

// ---------------------------------------------------------------------------
// GEMM: OUT = (X(kM x kD) @ W(kD x kD) + bias) * oscale, all-bf16 operands.
// Block = 128(M) x 64(N), 8 waves, wave = 16x64 strip.  W tile is staged into
// LDS *transposed* [n][kk] (stride 40 halfwords: bank-friendly, 16B-aligned)
// so B-fragments are 2x ds_load_b128.
// LAYOUT: 0 = bf16 [B,H,S,Dk] (Q,K);  1 = bf16 [B,H,Dk,S] (V^T);  2 = fp32 plain.
// ---------------------------------------------------------------------------
constexpr int kWT = 40;  // LDS row stride (halfwords) for 32 K-values + pad

template <int LAYOUT>
__global__ void __launch_bounds__(256)
mha_gemm_kernel(const __bf16* __restrict__ X, const __bf16* __restrict__ W,
                const float* __restrict__ bias, void* __restrict__ outp,
                float oscale) {
  __shared__ alignas(16) __bf16 ldsWt[64 * kWT];
  const int lane = threadIdx.x & 31;
  const int wid  = threadIdx.x >> 5;
  const int m0   = blockIdx.y * 128 + wid * 16;
  const int n0   = blockIdx.x * 64;

  v8f acc[4] = {v8zero(), v8zero(), v8zero(), v8zero()};

  const int sn  = threadIdx.x & 63;        // 0..63 : column within tile
  const int sk0 = (threadIdx.x >> 6) * 8;  // 0,8,16,24 : k-run start

  for (int k0 = 0; k0 < kD; k0 += 32) {
    __syncthreads();
    bf8v wv;
#pragma unroll
    for (int j = 0; j < 8; ++j)  // coalesced row reads, transposed LDS write
      wv[j] = W[(size_t)(k0 + sk0 + j) * kD + n0 + sn];
    *(bf8v*)(ldsWt + (size_t)sn * kWT + sk0) = wv;
    __syncthreads();

    const v16bf a = load_a_frag(X + (size_t)m0 * kD + k0, kD, lane);
#pragma unroll
    for (int dt = 0; dt < 4; ++dt) {
      const v16bf b = load_b_frag_kcontig(ldsWt + dt * 16 * kWT, kWT, lane);
      acc[dt] = wmma_bf16(a, b, acc[dt]);
    }
  }

  const int half = lane >> 4;
  const int col  = lane & 15;
#pragma unroll
  for (int dt = 0; dt < 4; ++dt) {
#pragma unroll
    for (int e = 0; e < 8; ++e) {
      const int m = m0 + e + 8 * half;
      const int n = n0 + dt * 16 + col;
      const float val = (acc[dt][e] + bias[n]) * oscale;
      if constexpr (LAYOUT == 0) {         // [B,H,S,Dk] bf16
        __bf16* out = (__bf16*)outp;
        const int bb = m >> 11, s = m & (kS - 1);
        const int h  = n >> 6,  d = n & (kDk - 1);
        out[((size_t)(bb * kH + h) * kS + s) * kDk + d] = (__bf16)val;
      } else if constexpr (LAYOUT == 1) {  // [B,H,Dk,S] bf16 (V transposed)
        __bf16* out = (__bf16*)outp;
        const int bb = m >> 11, s = m & (kS - 1);
        const int h  = n >> 6,  d = n & (kDk - 1);
        out[((size_t)(bb * kH + h) * kDk + d) * kS + s] = (__bf16)val;
      } else {                             // fp32 row-major (final output)
        float* out = (float*)outp;
        out[(size_t)m * kD + n] = val;
      }
    }
  }
}

// ---------------------------------------------------------------------------
// Flash attention: one wave per (batch, head, 16-query block).
// Q,K in [B,H,S,Dk]; V in [B,H,Dk,S] so every fragment load is b128.
// Q is pre-scaled by log2(e)/sqrt(Dk), so softmax runs natively in exp2:
// exp2(s*log2e - m*log2e) == exp(s - m), and v_exp_f32 needs no pre-multiply.
// ---------------------------------------------------------------------------
constexpr int kPT = 40;  // LDS prob-tile row stride (halfwords)

__global__ void __launch_bounds__(128)
mha_attn_kernel(const __bf16* __restrict__ Q, const __bf16* __restrict__ K,
                const __bf16* __restrict__ V, __bf16* __restrict__ C) {
  __shared__ alignas(16) __bf16 ldsP[4 * 16 * kPT];
  const int lane = threadIdx.x & 31;
  const int wid  = threadIdx.x >> 5;
  const int half = lane >> 4;
  const int col  = lane & 15;
  const int qb = blockIdx.x * 4 + wid;
  const int h  = blockIdx.y;
  const int b  = blockIdx.z;
  const int q0 = qb * 16;

  __bf16* myP = ldsP + wid * (16 * kPT);

  const size_t headoff = (size_t)(b * kH + h) * kS * kDk;
  const __bf16* Qp = Q + headoff + (size_t)q0 * kDk;  // [s][d]
  const __bf16* Kp = K + headoff;                     // [s][d]
  const __bf16* Vp = V + headoff;                     // [d][s]

  const v16bf qa0 = load_a_frag(Qp, kDk, lane);       // d 0..31
  const v16bf qa1 = load_a_frag(Qp + 32, kDk, lane);  // d 32..63

  v8f acc[4] = {v8zero(), v8zero(), v8zero(), v8zero()};
  float mrun[8], lrun[8];
#pragma unroll
  for (int e = 0; e < 8; ++e) { mrun[e] = -1.0e30f; lrun[e] = 0.f; }

  auto do_tile = [&](int kbase, bool masked) {
    // prefetch next key tile (K rows + V^T rows)
    if (kbase + 32 + lane < kS) {
      __builtin_prefetch(Kp + (size_t)(kbase + 32 + lane) * kDk, 0, 0);
      __builtin_prefetch(Vp + (size_t)(lane << 1) * kS + kbase + 32, 0, 0);
    }

    // S tile = Q(16x64) @ K^T(64x32) as two 16x16 tiles (K-dim chained 32+32)
    v8f st[2];
#pragma unroll
    for (int t = 0; t < 2; ++t) {
      const __bf16* kt = Kp + (size_t)(kbase + t * 16) * kDk;
      const v16bf kb0 = load_b_frag_kcontig(kt, kDk, lane);       // d 0..31
      const v16bf kb1 = load_b_frag_kcontig(kt + 32, kDk, lane);  // d 32..63
      st[t] = wmma_bf16(qa1, kb1, wmma_bf16(qa0, kb0, v8zero()));
    }

    if (masked) {  // wave-uniform: only diagonal tiles pay for this
#pragma unroll
      for (int e = 0; e < 8; ++e) {
        const int qg = q0 + e + 8 * half;
#pragma unroll
        for (int t = 0; t < 2; ++t) {
          const int kg = kbase + t * 16 + col;
          st[t][e] = (kg <= qg) ? st[t][e] : -1.0e9f;
        }
      }
    }

    // online softmax in exp2 domain, rows in C layout (raw v_max: no canonicalize)
#pragma unroll
    for (int e = 0; e < 8; ++e) {
      const float rowm  = red_max16(fmax_raw(st[0][e], st[1][e]));
      const float mnew  = fmax_raw(mrun[e], rowm);
      const float alpha = __builtin_exp2f(mrun[e] - mnew);
      mrun[e] = mnew;
      const float p0 = __builtin_exp2f(st[0][e] - mnew);
      const float p1 = __builtin_exp2f(st[1][e] - mnew);
      lrun[e] = lrun[e] * alpha + red_sum16(p0 + p1);
      st[0][e] = p0;
      st[1][e] = p1;
#pragma unroll
      for (int dt = 0; dt < 4; ++dt) acc[dt][e] *= alpha;
    }

    // C-layout -> A-layout transit through per-wave LDS (DS in-order per wave)
#pragma unroll
    for (int t = 0; t < 2; ++t)
#pragma unroll
      for (int e = 0; e < 8; ++e)
        myP[(e + 8 * half) * kPT + t * 16 + col] = (__bf16)st[t][e];
    asm volatile("s_wait_dscnt 0" ::: "memory");

    const v16bf pa = load_a_frag(myP, kPT, lane);  // 2x ds_load_b128
#pragma unroll
    for (int dt = 0; dt < 4; ++dt) {
      const v16bf vb =
          load_b_frag_kcontig(Vp + (size_t)(dt * 16) * kS + kbase, kS, lane);
      acc[dt] = wmma_bf16(pa, vb, acc[dt]);  // ctx += P(16x32) @ V(32x16)
    }
  };

  const int nkb   = (q0 + 15) / 32 + 1;  // tiles needed under causal mask
  const int nfull = (q0 + 1) >> 5;       // tiles fully below the diagonal
  for (int kb = 0; kb < nfull; ++kb) do_tile(kb * 32, false);
  for (int kb = nfull; kb < nkb; ++kb) do_tile(kb * 32, true);

  // normalize; write ctx bf16 in [B, S, H*Dk] for the output projection
#pragma unroll
  for (int e = 0; e < 8; ++e) {
    const float inv = __builtin_amdgcn_rcpf(lrun[e]);
    const int qg = q0 + e + 8 * half;
#pragma unroll
    for (int dt = 0; dt < 4; ++dt)
      C[((size_t)(b * kS) + qg) * kD + h * kDk + dt * 16 + col] =
          (__bf16)(acc[dt][e] * inv);
  }
}

extern "C" void kernel_launch(void* const* d_in, const int* in_sizes, int n_in,
                              void* d_out, int out_size, void* d_ws, size_t ws_size,
                              hipStream_t stream) {
  (void)in_sizes; (void)n_in; (void)out_size; (void)ws_size;
  const float* query = (const float*)d_in[0];
  const float* key_i = (const float*)d_in[1];
  const float* value = (const float*)d_in[2];
  // d_in[3] is the causal mask; causality is applied analytically in-kernel.
  const float* wq = (const float*)d_in[4];
  const float* bq = (const float*)d_in[5];
  const float* wk = (const float*)d_in[6];
  const float* bk = (const float*)d_in[7];
  const float* wv = (const float*)d_in[8];
  const float* bv = (const float*)d_in[9];
  const float* wo = (const float*)d_in[10];
  const float* bo = (const float*)d_in[11];

  const size_t nX = (size_t)kM * kD;  // 4M elements
  const size_t nW = (size_t)kD * kD;  // 1M elements
  __bf16* p   = (__bf16*)d_ws;
  __bf16* Xq  = p; p += nX;   // bf16 copies of activations
  __bf16* Xk  = p; p += nX;
  __bf16* Xv  = p; p += nX;
  __bf16* Wqb = p; p += nW;   // bf16 copies of weights
  __bf16* Wkb = p; p += nW;
  __bf16* Wvb = p; p += nW;
  __bf16* Wob = p; p += nW;
  __bf16* Qw  = p; p += nX;   // [B,H,S,Dk], pre-scaled by log2e/8
  __bf16* Kw  = p; p += nX;   // [B,H,S,Dk]
  __bf16* Vw  = p; p += nX;   // [B,H,Dk,S] (transposed)
  __bf16* Cw  = p;            // ctx [B,S,D]

  const int cb = 256 * 8;
  cvt_bf16_kernel<<<nX / cb, 256, 0, stream>>>(query, Xq, (int)nX);
  cvt_bf16_kernel<<<nX / cb, 256, 0, stream>>>(key_i, Xk, (int)nX);
  cvt_bf16_kernel<<<nX / cb, 256, 0, stream>>>(value, Xv, (int)nX);
  cvt_bf16_kernel<<<nW / cb, 256, 0, stream>>>(wq, Wqb, (int)nW);
  cvt_bf16_kernel<<<nW / cb, 256, 0, stream>>>(wk, Wkb, (int)nW);
  cvt_bf16_kernel<<<nW / cb, 256, 0, stream>>>(wv, Wvb, (int)nW);
  cvt_bf16_kernel<<<nW / cb, 256, 0, stream>>>(wo, Wob, (int)nW);

  const float kQScale = 0.125f * 1.44269504088896340736f;  // log2e / sqrt(Dk)
  dim3 ggrid(kD / 64, kM / 128);  // (16, 32)
  mha_gemm_kernel<0><<<ggrid, 256, 0, stream>>>(Xq, Wqb, bq, Qw, kQScale);
  mha_gemm_kernel<0><<<ggrid, 256, 0, stream>>>(Xk, Wkb, bk, Kw, 1.0f);
  mha_gemm_kernel<1><<<ggrid, 256, 0, stream>>>(Xv, Wvb, bv, Vw, 1.0f);

  mha_attn_kernel<<<dim3(kS / 16 / 4, kH, kB), 128, 0, stream>>>(Qw, Kw, Vw, Cw);

  mha_gemm_kernel<2><<<ggrid, 256, 0, stream>>>(Cw, Wob, bo, d_out, 1.0f);
}
